// BiTransformer_layer_74801150427546
// MI455X (gfx1250) — compile-verified, matrix operand order checked by reading, not run on hardware
//
#include <hip/hip_runtime.h>

#define BB 2
#define SS 2048
#define DD 768
#define HH 12
#define DHH 64
#define DVV 32
#define DFF_ 3072
#define MM (BB*SS)          // 4096 rows
#define MASK_VAL (-10000.0f)

typedef __bf16 v16bf __attribute__((ext_vector_type(16)));
typedef __bf16 v8bf  __attribute__((ext_vector_type(8)));
typedef float  v8f   __attribute__((ext_vector_type(8)));
typedef unsigned int u32x4 __attribute__((ext_vector_type(4)));
typedef int i32x4 __attribute__((ext_vector_type(4)));
typedef int i32x8 __attribute__((ext_vector_type(8)));

#if __has_builtin(__builtin_amdgcn_tensor_load_to_lds) && __has_builtin(__builtin_amdgcn_s_wait_tensorcnt)
#define USE_TDM 1
#else
#define USE_TDM 0
#endif

// Element index -> K offset inside a 16-bit WMMA A/B fragment:
// lanes 0-15 hold K blocks {0..7, 16..23}; lanes 16-31 hold {8..15, 24..31}.
__device__ __forceinline__ int kidx(int e, int half) {
    return ((e >> 3) << 4) + (half << 3) + (e & 7);
}
__device__ __forceinline__ v8bf ld8(const __bf16* p) { return *(const v8bf*)p; }

union V16U { v16bf v; v8bf h[2]; };

// Build a 16-element bf16 fragment from a 32-element row: chunks [half] and [2+half].
__device__ __forceinline__ v16bf frag_from_row(const __bf16* row32, int half) {
    V16U u;
    const v8bf* p = (const v8bf*)row32;
    u.h[0] = p[half];
    u.h[1] = p[2 + half];
    return u.v;
}

// ---------------------------------------------------------------------------
// f32 -> bf16 conversion
// ---------------------------------------------------------------------------
__global__ __launch_bounds__(256) void cvt_bf16_kernel(const float* __restrict__ in,
                                                       __bf16* __restrict__ out, int n) {
    int i = blockIdx.x * 256 + threadIdx.x;
    if (i < n) out[i] = (__bf16)in[i];
}

// ---------------------------------------------------------------------------
// bf16 WMMA GEMM, double-buffered LDS staging. Block = 256 threads (8 waves),
// tile 128(M)x64(N), K-step 32. Waves: 4 in M x 2 in N, each wave 2x2 WMMA
// tiles. A tile (128x32) DMA'd by the Tensor Data Mover into the next buffer
// while the current buffer feeds WMMAs; W tile staged transposed so both
// fragments are contiguous b128 LDS reads.
// ---------------------------------------------------------------------------
template <bool LEAKY, bool OUTF, bool OUTB>
__global__ __launch_bounds__(256) void gemm_wmma_kernel(
        const __bf16* __restrict__ A, const __bf16* __restrict__ W,
        const float* __restrict__ bias, float* __restrict__ Cf,
        __bf16* __restrict__ Cb, int M, int N, int K) {
    extern __shared__ char smem[];
    // LDS layout: A buffers at 0 / 8K ([128][32] each); W^T buffers at 16K / 20K.
    auto lds_a  = [&](int b) -> __bf16* { return (__bf16*)(smem + (b << 13)); };
    auto lds_wt = [&](int b) -> __bf16* { return (__bf16*)(smem + 16384 + (b << 12)); };

    const int tid  = threadIdx.x;
    const int lane = tid & 31;
    const int wid  = tid >> 5;
    const int wm   = wid & 3;        // wave row (4 in M)
    const int wn   = wid >> 2;       // wave col (2 in N)
    const int mr   = lane & 15;
    const int half = lane >> 4;
    const int m0   = blockIdx.y * 128;
    const int n0   = blockIdx.x * 64;
    const int kr   = tid >> 3;       // W stage: row 0..31
    const int c0   = (tid & 7) * 8;  // W stage: col chunk

#if USE_TDM
    auto tdm_issue = [&](int kk, int buf) {
        // D# group0: count=1, lds_addr = A-buffer offset, tile global addr, type=2
        unsigned long long gaddr =
            (unsigned long long)(const void*)(A + (size_t)m0 * K + kk);
        u32x4 g0;
        g0[0] = 1u;
        g0[1] = (unsigned)(buf << 13);
        g0[2] = (unsigned)gaddr;
        g0[3] = (unsigned)(gaddr >> 32) | (2u << 30);
        // D# group1: data_size=2B, tensor(K x M), tile(32 x 128), stride0=K
        i32x8 g1;
        g1[0] = 1 << 16;                                   // data_size=1 -> 2B
        g1[1] = (K & 0xFFFF) << 16;                        // tensor_dim0 lo
        g1[2] = (K >> 16) | ((M & 0xFFFF) << 16);          // dim0 hi | dim1 lo
        g1[3] = (M >> 16) | (32 << 16);                    // dim1 hi | tile_dim0
        g1[4] = 128;                                       // tile_dim1
        g1[5] = K;                                         // dim0_stride lo
        g1[6] = 0;
        g1[7] = 0;
        i32x4 gz4 = {};
        i32x8 gz8 = {};
        __builtin_amdgcn_tensor_load_to_lds(g0, g1, gz4, gz4, gz8, 0);
    };
#else
    auto coop_a_load = [&](int kk, v8bf* dst) {
#pragma unroll
        for (int i = 0; i < 2; ++i) {
            const int c = tid + i * 256;
            const int row = c >> 2, cc = c & 3;
            dst[i] = ld8(A + (size_t)(m0 + row) * K + kk + cc * 8);
        }
    };
    auto coop_a_store = [&](int buf, const v8bf* src) {
#pragma unroll
        for (int i = 0; i < 2; ++i)
            ((v8bf*)lds_a(buf))[tid + i * 256] = src[i];
    };
#endif
    auto w_store = [&](int buf, v8bf wrow) {
        __bf16* wt = lds_wt(buf);
#pragma unroll
        for (int e = 0; e < 8; ++e)
            wt[(c0 + e) * 32 + kr] = wrow[e];
    };

    v8f acc[2][2] = {};

    // ---- prologue: stage k=0 into buffer 0 ----
#if USE_TDM
    if (wid == 0) tdm_issue(0, 0);
#else
    {
        v8bf a0[2];
        coop_a_load(0, a0);
        coop_a_store(0, a0);
    }
#endif
    w_store(0, ld8(W + (size_t)kr * N + n0 + c0));
#if USE_TDM
    if (wid == 0) __builtin_amdgcn_s_wait_tensorcnt(0);
#endif
    __syncthreads();

    int buf = 0;
    for (int k0 = 0; k0 < K; k0 += 32) {
        const int knext = k0 + 32;
        const bool more = knext < K;

        // ---- kick off next tile (TDM DMA + W global loads) ----
        v8bf wnext = {};
#if USE_TDM
        if (more && wid == 0) tdm_issue(knext, buf ^ 1);
#else
        v8bf anext[2];
        if (more) coop_a_load(knext, anext);
#endif
        if (more) wnext = ld8(W + (size_t)(knext + kr) * N + n0 + c0);

        // ---- WMMAs on current buffer (overlaps the DMA) ----
        const __bf16* la = lds_a(buf);
        const __bf16* lw = lds_wt(buf);
        v16bf af[2], bf_[2];
#pragma unroll
        for (int mt = 0; mt < 2; ++mt)
            af[mt] = frag_from_row(&la[(wm * 32 + mt * 16 + mr) * 32], half);
#pragma unroll
        for (int nt = 0; nt < 2; ++nt)
            bf_[nt] = frag_from_row(&lw[(wn * 32 + nt * 16 + mr) * 32], half);
#pragma unroll
        for (int mt = 0; mt < 2; ++mt)
#pragma unroll
            for (int nt = 0; nt < 2; ++nt)
                acc[mt][nt] = __builtin_amdgcn_wmma_f32_16x16x32_bf16(
                        false, af[mt], false, bf_[nt], (short)0, acc[mt][nt],
                        false, false);

        // ---- commit next tile's staging, wait for DMA, flip ----
        if (more) {
            w_store(buf ^ 1, wnext);
#if !USE_TDM
            coop_a_store(buf ^ 1, anext);
#endif
#if USE_TDM
            if (wid == 0) __builtin_amdgcn_s_wait_tensorcnt(0);
#endif
        }
        __syncthreads();
        buf ^= 1;
    }

#pragma unroll
    for (int mt = 0; mt < 2; ++mt)
#pragma unroll
        for (int nt = 0; nt < 2; ++nt) {
            const int col = n0 + wn * 32 + nt * 16 + mr;
            const float bsc = bias[col];
#pragma unroll
            for (int r = 0; r < 8; ++r) {
                const int row = m0 + wm * 32 + mt * 16 + r + 8 * half;
                float v = acc[mt][nt][r] + bsc;
                if (LEAKY) v = v > 0.0f ? v : 0.01f * v;
                const size_t off = (size_t)row * N + col;
                if (OUTF) Cf[off] = v;
                if (OUTB) Cb[off] = (__bf16)v;
            }
        }
}

// ---------------------------------------------------------------------------
// Bidirectional flash attention. One wave per (b, h, 16-row tile).
// dir 0: look-ahead (keep j<=i), dir 1: look-behind (keep j>=i).
// Only the single diagonal 32-wide j-tile per direction is masked.
// ---------------------------------------------------------------------------
__global__ __launch_bounds__(32) void attn_bidir_kernel(
        const __bf16* __restrict__ qb, const __bf16* __restrict__ kb,
        const __bf16* __restrict__ vb, __bf16* __restrict__ cat) {
    __shared__ __bf16 lds_pb[16 * 32];
    __shared__ __bf16 lds_v[32 * 32];

    const int lane = threadIdx.x;
    const int it = blockIdx.x, h = blockIdx.y, b = blockIdx.z;
    const int i0 = it * 16;
    const int mr = lane & 15;
    const int half = lane >> 4;

    // Q fragments: 16x64 = two 16x32 A-fragments, contiguous b128 chunks
    v16bf qa[2];
    const __bf16* qrow = qb + ((size_t)(b * SS + i0 + mr)) * DD + h * DHH;
#pragma unroll
    for (int c = 0; c < 2; ++c) {
        V16U u;
        u.h[0] = ld8(qrow + c * 32 + half * 8);
        u.h[1] = ld8(qrow + c * 32 + 16 + half * 8);
        qa[c] = u.v;
    }

    const int jd = i0 & ~31;   // the only masked j-tile per direction

    for (int dir = 0; dir < 2; ++dir) {
        const bool fwd = (dir == 0);
        float rm[8], rs[8];
        v8f acc0 = {}, acc1 = {};
#pragma unroll
        for (int r = 0; r < 8; ++r) { rm[r] = -3.0e38f; rs[r] = 0.0f; }

        // mode: 0 = no mask, 1 = fwd diagonal (keep j<=i), 2 = bwd diagonal (keep j>=i)
        auto step = [&](int j0, int mode) {
            // ---- stage V tile (32 rows x 32 cols) into LDS: 4 b128 per lane ----
            const __bf16* vrow =
                vb + ((size_t)(b * SS + j0 + lane)) * (HH * DVV) + h * DVV;
#pragma unroll
            for (int c = 0; c < 4; ++c)
                ((v8bf*)lds_v)[lane * 4 + c] = ld8(vrow + c * 8);

            // ---- scores: two 16-col sub-tiles, d=64 split as 2x32 ----
            v8f sc[2];
#pragma unroll
            for (int t = 0; t < 2; ++t) {
                v8f c = {};
#pragma unroll
                for (int kc = 0; kc < 2; ++kc) {
                    const __bf16* krow =
                        kb + ((size_t)(b * SS + j0 + t * 16 + mr)) * DD + h * DHH + kc * 32;
                    V16U u;
                    u.h[0] = ld8(krow + half * 8);
                    u.h[1] = ld8(krow + 16 + half * 8);
                    c = __builtin_amdgcn_wmma_f32_16x16x32_bf16(
                            false, qa[kc], false, u.v, (short)0, c, false, false);
                }
                sc[t] = c;
            }
            // ---- scale (+ diagonal mask) + tile row-max ----
            float sv[2][8], pmax[8];
#pragma unroll
            for (int r = 0; r < 8; ++r) pmax[r] = -3.0e38f;
#pragma unroll
            for (int t = 0; t < 2; ++t)
#pragma unroll
                for (int r = 0; r < 8; ++r) {
                    float s = sc[t][r] * 0.125f;   // 1/sqrt(64)
                    if (mode != 0) {
                        const int i = i0 + r + 8 * half;
                        const int j = j0 + t * 16 + mr;
                        const bool keep = (mode == 1) ? (j <= i) : (j >= i);
                        s = keep ? s : MASK_VAL;
                    }
                    sv[t][r] = s;
                    pmax[r] = fmaxf(pmax[r], s);
                }
#pragma unroll
            for (int r = 0; r < 8; ++r)
                for (int off = 8; off; off >>= 1)
                    pmax[r] = fmaxf(pmax[r], __shfl_xor(pmax[r], off, 16));
            // ---- online softmax rescale ----
            float psum[8];
#pragma unroll
            for (int r = 0; r < 8; ++r) {
                const float mnew = fmaxf(rm[r], pmax[r]);
                const float scl = __expf(rm[r] - mnew);
                rm[r] = mnew;
                rs[r] *= scl;
                acc0[r] *= scl;
                acc1[r] *= scl;
                psum[r] = 0.0f;
            }
#pragma unroll
            for (int t = 0; t < 2; ++t)
#pragma unroll
                for (int r = 0; r < 8; ++r) {
                    const float p = __expf(sv[t][r] - rm[r]);
                    psum[r] += p;
                    lds_pb[(r + 8 * half) * 32 + t * 16 + mr] = (__bf16)p;
                }
#pragma unroll
            for (int r = 0; r < 8; ++r) {
                for (int off = 8; off; off >>= 1)
                    psum[r] += __shfl_xor(psum[r], off, 16);
                rs[r] += psum[r];
            }
            __syncthreads();
            // ---- P back as contiguous A-fragment ----
            v16bf pa;
            {
                V16U u;
                const v8bf* lp = (const v8bf*)lds_pb;
                u.h[0] = lp[mr * 4 + half];
                u.h[1] = lp[mr * 4 + 2 + half];
                pa = u.v;
            }
            // ---- PV: A = P (16x32), B = V (32x16) per DV half ----
#pragma unroll
            for (int t = 0; t < 2; ++t) {
                v16bf vfrag;
#pragma unroll
                for (int e = 0; e < 16; ++e)
                    vfrag[e] = lds_v[kidx(e, half) * 32 + t * 16 + mr];
                if (t == 0)
                    acc0 = __builtin_amdgcn_wmma_f32_16x16x32_bf16(
                            false, pa, false, vfrag, (short)0, acc0, false, false);
                else
                    acc1 = __builtin_amdgcn_wmma_f32_16x16x32_bf16(
                            false, pa, false, vfrag, (short)0, acc1, false, false);
            }
            __syncthreads();
        };

        if (fwd) {
            for (int j0 = 0; j0 < jd; j0 += 32) step(j0, 0);   // interior, unmasked
            step(jd, 1);                                        // diagonal
        } else {
            step(jd, 2);                                        // diagonal
            for (int j0 = jd + 32; j0 < SS; j0 += 32) step(j0, 0);
        }

        // ---- normalize and store ----
        const int base = fwd ? (h * DVV) : (HH * DVV + h * DVV);
#pragma unroll
        for (int r = 0; r < 8; ++r) {
            const int i = i0 + r + 8 * half;
            const float inv = 1.0f / rs[r];
            const size_t o = ((size_t)(b * SS + i)) * DD + base + mr;
            cat[o]      = (__bf16)(acc0[r] * inv);
            cat[o + 16] = (__bf16)(acc1[r] * inv);
        }
    }
}

// ---------------------------------------------------------------------------
// y = LayerNorm(a + b) * gamma + beta ; writes f32 and optionally bf16.
// ---------------------------------------------------------------------------
template <bool STOREB>
__global__ __launch_bounds__(256) void add_ln_kernel(
        const float* __restrict__ a, const float* __restrict__ bsrc,
        const float* __restrict__ gamma, const float* __restrict__ beta,
        float* __restrict__ outf, __bf16* __restrict__ outb) {
    const int row = blockIdx.x;
    const float* ar = a + (size_t)row * DD;
    const float* br = bsrc + (size_t)row * DD;
    float v[3], s = 0.0f, s2 = 0.0f;
#pragma unroll
    for (int t = 0; t < 3; ++t) {
        const float x = ar[threadIdx.x + t * 256] + br[threadIdx.x + t * 256];
        v[t] = x; s += x; s2 += x * x;
    }
    for (int off = 16; off; off >>= 1) {
        s  += __shfl_xor(s,  off, 32);
        s2 += __shfl_xor(s2, off, 32);
    }
    __shared__ float sh[8], sh2[8], mv[2];
    const int w = threadIdx.x >> 5;
    if ((threadIdx.x & 31) == 0) { sh[w] = s; sh2[w] = s2; }
    __syncthreads();
    if (threadIdx.x == 0) {
        float ts = 0.0f, ts2 = 0.0f;
        for (int i = 0; i < 8; ++i) { ts += sh[i]; ts2 += sh2[i]; }
        const float mean = ts * (1.0f / DD);
        const float var  = ts2 * (1.0f / DD) - mean * mean;
        mv[0] = mean;
        mv[1] = rsqrtf(var + 1e-5f);
    }
    __syncthreads();
    const float mean = mv[0], rstd = mv[1];
#pragma unroll
    for (int t = 0; t < 3; ++t) {
        const int c = threadIdx.x + t * 256;
        const float y = (v[t] - mean) * rstd * gamma[c] + beta[c];
        outf[(size_t)row * DD + c] = y;
        if (STOREB) outb[(size_t)row * DD + c] = (__bf16)y;
    }
}

// ---------------------------------------------------------------------------
// Host launcher
// ---------------------------------------------------------------------------
extern "C" void kernel_launch(void* const* d_in, const int* in_sizes, int n_in,
                              void* d_out, int out_size, void* d_ws, size_t ws_size,
                              hipStream_t stream) {
    const float* x    = (const float*)d_in[0];
    const float* wq   = (const float*)d_in[1];
    const float* bq   = (const float*)d_in[2];
    const float* wk   = (const float*)d_in[3];
    const float* bk   = (const float*)d_in[4];
    const float* wv   = (const float*)d_in[5];
    const float* bv   = (const float*)d_in[6];
    const float* wo   = (const float*)d_in[7];
    const float* bo   = (const float*)d_in[8];
    const float* w1   = (const float*)d_in[9];
    const float* b1   = (const float*)d_in[10];
    const float* w2   = (const float*)d_in[11];
    const float* b2   = (const float*)d_in[12];
    const float* ln1s = (const float*)d_in[13];
    const float* ln1b = (const float*)d_in[14];
    const float* ln2s = (const float*)d_in[15];
    const float* ln2b = (const float*)d_in[16];
    float* out = (float*)d_out;

    char* ws = (char*)d_ws;
    size_t off = 0;
    auto alloc = [&](size_t bytes) -> void* {
        off = (off + 255) & ~(size_t)255;
        void* p = ws + off;
        off += bytes;
        return p;
    };

    __bf16* XB  = (__bf16*)alloc((size_t)MM * DD * 2);
    __bf16* WQB = (__bf16*)alloc((size_t)DD * DD * 2);
    __bf16* WKB = (__bf16*)alloc((size_t)DD * DD * 2);
    __bf16* WVB = (__bf16*)alloc((size_t)DD * (DD / 2) * 2);
    __bf16* WOB = (__bf16*)alloc((size_t)DD * DD * 2);
    __bf16* W1B = (__bf16*)alloc((size_t)DD * DFF_ * 2);
    __bf16* W2B = (__bf16*)alloc((size_t)DFF_ * DD * 2);
    __bf16* QB  = (__bf16*)alloc((size_t)MM * DD * 2);
    __bf16* KB  = (__bf16*)alloc((size_t)MM * DD * 2);
    __bf16* VB  = (__bf16*)alloc((size_t)MM * (DD / 2) * 2);
    __bf16* CAT = (__bf16*)alloc((size_t)MM * DD * 2);
    float*  AO  = (float*)alloc((size_t)MM * DD * 4);
    float*  H1F = (float*)alloc((size_t)MM * DD * 4);
    __bf16* H1B = (__bf16*)alloc((size_t)MM * DD * 2);
    __bf16* F1B = (__bf16*)alloc((size_t)MM * DFF_ * 2);
    float*  F2F = (float*)alloc((size_t)MM * DD * 4);

    auto cvt = [&](const float* src, __bf16* dst, int n) {
        cvt_bf16_kernel<<<(n + 255) / 256, 256, 0, stream>>>(src, dst, n);
    };
    cvt(x,  XB,  MM * DD);
    cvt(wq, WQB, DD * DD);
    cvt(wk, WKB, DD * DD);
    cvt(wv, WVB, DD * (DD / 2));
    cvt(wo, WOB, DD * DD);
    cvt(w1, W1B, DD * DFF_);
    cvt(w2, W2B, DFF_ * DD);

    const int LDS_BYTES = 24576;   // double-buffered A (2x8K) + W^T (2x4K)

    // Q, K, V projections (bf16 outputs)
    gemm_wmma_kernel<false, false, true>
        <<<dim3(DD / 64, MM / 128), 256, LDS_BYTES, stream>>>(
            XB, WQB, bq, nullptr, QB, MM, DD, DD);
    gemm_wmma_kernel<false, false, true>
        <<<dim3(DD / 64, MM / 128), 256, LDS_BYTES, stream>>>(
            XB, WKB, bk, nullptr, KB, MM, DD, DD);
    gemm_wmma_kernel<false, false, true>
        <<<dim3((DD / 2) / 64, MM / 128), 256, LDS_BYTES, stream>>>(
            XB, WVB, bv, nullptr, VB, MM, DD / 2, DD);

    // Bidirectional flash attention -> CAT (bf16)
    attn_bidir_kernel<<<dim3(SS / 16, HH, BB), 32, 0, stream>>>(QB, KB, VB, CAT);

    // Output projection (f32)
    gemm_wmma_kernel<false, true, false>
        <<<dim3(DD / 64, MM / 128), 256, LDS_BYTES, stream>>>(
            CAT, WOB, bo, AO, nullptr, MM, DD, DD);

    // h1 = LN(x + attn_out)
    add_ln_kernel<true><<<MM, 256, 0, stream>>>(x, AO, ln1s, ln1b, H1F, H1B);

    // FFN
    gemm_wmma_kernel<true, false, true>
        <<<dim3(DFF_ / 64, MM / 128), 256, LDS_BYTES, stream>>>(
            H1B, W1B, b1, nullptr, F1B, MM, DFF_, DD);
    gemm_wmma_kernel<false, true, false>
        <<<dim3(DD / 64, MM / 128), 256, LDS_BYTES, stream>>>(
            F1B, W2B, b2, F2F, nullptr, MM, DD, DFF_);

    // out = LN(h1 + ffn)
    add_ln_kernel<false><<<MM, 256, 0, stream>>>(H1F, F2F, ln2s, ln2b, out, nullptr);
}
